// HybridQLSTM_65481071401132
// MI455X (gfx1250) — matrix operand order repaired and checked by reading, not compiled, not run on hardware
//
#include <hip/hip_runtime.h>
#include <hip/hip_bf16.h>

// ---------------------------------------------------------------------------
// Types
// ---------------------------------------------------------------------------
typedef __bf16 bf16_t;
typedef bf16_t v16bf __attribute__((ext_vector_type(16)));
typedef float  v8f   __attribute__((ext_vector_type(8)));
typedef unsigned int  u32x4 __attribute__((ext_vector_type(4)));
typedef int           s32x4 __attribute__((ext_vector_type(4)));
typedef int           s32x8 __attribute__((ext_vector_type(8)));
typedef unsigned short bf16raw;

#define VOCAB 32000
#define EMB 512
#define HID 512
#define TAGS 64
#define BATCH 32
#define SEQ 512
#define DIN 1024          // EMB + HID
#define NOUT 2048         // 4 * HID
#define NWG 16            // workgroups in recurrent scan (one WGP each)
#define SCAN_THREADS 128  // 4 waves per scan WG

// ---------------------------------------------------------------------------
// Helpers
// ---------------------------------------------------------------------------
__device__ __forceinline__ bf16raw f2bf(float f) {
  unsigned int u = __float_as_uint(f);
  unsigned int r = (u + 0x7fffu + ((u >> 16) & 1u)) >> 16; // RNE
  return (bf16raw)r;
}

__device__ __forceinline__ v8f wmma_bf16(v16bf a, v16bf b, v8f c) {
  // v_wmma_f32_16x16x32_bf16  D = A*B + C  (fp32 accumulate)
  return __builtin_amdgcn_wmma_f32_16x16x32_bf16(
      /*neg_a=*/false, a, /*neg_b=*/false, b,
      /*c_mod=*/(short)0, c, /*reuse_a=*/false, /*reuse_b=*/false);
}

// A-fragment union: 16x32 bf16 tile, per-lane halves are 16B contiguous.
union AFrag { v16bf v; u32x4 q[2]; };

// Load A-fragment from a row-major bf16 matrix (global or LDS).
// ISA layout: lane l (l<16): M=l, K = {0..7,16..23}; lane l+16: M=l,
// K={8..15,24..31}. Exactly 2 b128 loads per lane.
__device__ __forceinline__ void load_a_frag(AFrag& fr, const bf16raw* base,
                                            int row0, int kbase, int ld) {
  const int lane = threadIdx.x & 31;
  const int row  = row0 + (lane & 15);
  const int kb   = kbase + ((lane >> 4) << 3);
  const bf16raw* p = base + (size_t)row * ld + kb;
  fr.q[0] = *reinterpret_cast<const u32x4*>(p);
  fr.q[1] = *reinterpret_cast<const u32x4*>(p + 16);
}

// Tensor Data Mover: async 2D copy global -> LDS, dword-granular.
// xdw dwords per row (<= 65535), `rows` rows, row stride in dwords.
// D# built per CDNA5 ISA ch.8: group0 = flags/lds_addr/global_addr/type=2,
// group1 = data_size(4B)/dims/tile dims/strides. Tracked with TENSORcnt.
__device__ __forceinline__ void tdm_load_2d(const void* lds_ptr,
                                            const void* gptr,
                                            unsigned int xdw,
                                            unsigned int rows,
                                            unsigned int stride_dw) {
  const unsigned long long ga = (unsigned long long)(size_t)gptr;
  const unsigned int lds_off = (unsigned int)(size_t)lds_ptr; // flat low32 = LDS byte addr
  u32x4 g0;
  g0[0] = 1u;                                      // count=1, user descriptor
  g0[1] = lds_off;                                 // lds_addr (bytes)
  g0[2] = (unsigned int)(ga & 0xffffffffu);        // global_addr[31:0]
  g0[3] = (unsigned int)((ga >> 32) & 0x01ffffffu) // global_addr[56:32]
          | (2u << 30);                            // type=2 ("image")
  s32x8 g1;
  g1[0] = (int)(2u << 16);                         // data_size = 4 bytes
  g1[1] = (int)((xdw & 0xffffu) << 16);            // tensor_dim0[15:0]
  g1[2] = (int)((xdw >> 16) | ((rows & 0xffffu) << 16));   // dim0 hi | dim1 lo
  g1[3] = (int)((rows >> 16) | ((xdw & 0xffffu) << 16));   // dim1 hi | tile_dim0
  g1[4] = (int)(rows & 0xffffu);                   // tile_dim1 (tile_dim2 = 0)
  g1[5] = (int)stride_dw;                          // tensor_dim0_stride lo32
  g1[6] = 0;
  g1[7] = 0;
  s32x4 g2 = {0, 0, 0, 0};
  s32x4 g3 = {0, 0, 0, 0};
#if defined(__clang_major__) && (__clang_major__ >= 23)
  s32x8 g4 = {0, 0, 0, 0, 0, 0, 0, 0};
  __builtin_amdgcn_tensor_load_to_lds(g0, g1, g2, g3, g4, 0);
#else
  __builtin_amdgcn_tensor_load_to_lds(g0, g1, g2, g3, 0);
#endif
}

__device__ __forceinline__ float sigm(float x) {
  return 1.0f / (1.0f + __expf(-x));
}

// ---------------------------------------------------------------------------
// Phase 0a: init barrier + zero h(0)
// ---------------------------------------------------------------------------
__global__ void init_kernel(bf16raw* Hseq, unsigned int* bar) {
  const int tid = blockIdx.x * 256 + threadIdx.x;
  if (tid == 0) *bar = 0u;
  if (tid < BATCH * HID) Hseq[tid] = 0; // h at t=0 is zero
}

// ---------------------------------------------------------------------------
// Phase 0b: one-time bf16 conversion of the embedding table.
// ---------------------------------------------------------------------------
__global__ void pack_e_kernel(const float* E, bf16raw* Ebf) {
  const size_t tid = (size_t)blockIdx.x * 256 + threadIdx.x;
  if (tid < (size_t)VOCAB * EMB) Ebf[tid] = f2bf(E[tid]);
}

// ---------------------------------------------------------------------------
// Phase 0c: pack W4 = [Wf|Wi|Wg|Wo] (1024x2048 fp32) into bf16 B-fragments.
// Fragment (kt,ntile): 32 lanes x 16 K values per lane, K = {0..7,16..23} (+8 for lanes 16..31).
// ---------------------------------------------------------------------------
__global__ void pack_w4_kernel(const float* Wf, const float* Wi,
                               const float* Wg, const float* Wo,
                               bf16raw* W4P) {
  const int tid = blockIdx.x * 256 + threadIdx.x; // 32*128*512 = 2097152
  const int e = tid & 15;
  const int lane = (tid >> 4) & 31;
  const int frag = tid >> 9;
  const int kt = frag >> 7;
  const int ntile = frag & 127;
  const int col = ntile * 16 + (lane & 15);
  const int krow = kt * 32 + ((lane >> 4) << 3) + (e & 7) + ((e >> 3) << 4);
  const float* W = (col < 512) ? Wf : (col < 1024) ? Wi : (col < 1536) ? Wg : Wo;
  W4P[tid] = f2bf(W[(size_t)krow * HID + (col & 511)]);
}

__global__ void pack_w2_kernel(const float* W2, bf16raw* W2P) {
  const int tid = blockIdx.x * 256 + threadIdx.x; // 16*4*512 = 32768
  const int e = tid & 15;
  const int lane = (tid >> 4) & 31;
  const int frag = tid >> 9;
  const int kt = frag >> 2;
  const int ntile = frag & 3;
  const int col = ntile * 16 + (lane & 15);
  const int krow = kt * 32 + ((lane >> 4) << 3) + (e & 7) + ((e >> 3) << 4);
  W2P[tid] = f2bf(W2[(size_t)krow * TAGS + col]);
}

// ---------------------------------------------------------------------------
// Phase 1: Zx[s*32+b, 0:2048] = embeds @ Wx + b4   (fully parallel WMMA GEMM)
// One wave owns (mtile, 4 N-tiles). Two-deep software pipeline with explicit
// double-buffered A/B fragments: buffer A is live while buffer B loads, so
// the allocator must keep them in distinct registers and the waits become
// partial instead of wait-0 per WMMA.
// ---------------------------------------------------------------------------
__global__ void __launch_bounds__(256)
zx_gemm_kernel(const int* __restrict__ sentence, const bf16raw* __restrict__ Ebf,
               const bf16raw* __restrict__ W4P, const float* bf_,
               const float* bi_, const float* bg_, const float* bo_,
               float* __restrict__ Zx) {
  const int wid = (blockIdx.x * 256 + threadIdx.x) >> 5; // 32768 waves
  const int lane = threadIdx.x & 31;
  const int mtile = wid >> 5;   // 1024 M-tiles (M = 16384 = S*B)
  const int ng = wid & 31;      // 32 groups of 4 N-tiles

  // Per-lane embedding row base (row = token for this fragment row).
  const int gr = mtile * 16 + (lane & 15);
  const int s = gr >> 5, b = gr & 31;
  const int tok = sentence[b * SEQ + s];
  const bf16raw* arow = Ebf + (size_t)tok * EMB + ((lane >> 4) << 3);
  const bf16raw* wbase = W4P + (size_t)(ng * 4) * 512 + (lane << 4);

  v8f acc[4];
#pragma unroll
  for (int j = 0; j < 4; ++j) acc[j] = (v8f){0,0,0,0,0,0,0,0};

  AFrag aA, aB;
  v16bf bA[4], bB[4];
  // Prologue: stage kt=0 into buffer A.
  aA.q[0] = *reinterpret_cast<const u32x4*>(arow);
  aA.q[1] = *reinterpret_cast<const u32x4*>(arow + 16);
#pragma unroll
  for (int j = 0; j < 4; ++j)
    bA[j] = *reinterpret_cast<const v16bf*>(wbase + (size_t)j * 512);

#pragma unroll
  for (int kt = 0; kt < 16; kt += 2) {
    // Stage kt+1 into buffer B while buffer A is consumed.
    const int k1 = kt + 1;
    aB.q[0] = *reinterpret_cast<const u32x4*>(arow + k1 * 32);
    aB.q[1] = *reinterpret_cast<const u32x4*>(arow + k1 * 32 + 16);
#pragma unroll
    for (int j = 0; j < 4; ++j)
      bB[j] = *reinterpret_cast<const v16bf*>(wbase +
                                              (size_t)(k1 * 128 + j) * 512);
#pragma unroll
    for (int j = 0; j < 4; ++j) acc[j] = wmma_bf16(aA.v, bA[j], acc[j]);

    // Stage kt+2 into buffer A while buffer B is consumed.
    const int k2 = kt + 2;
    if (k2 < 16) {
      aA.q[0] = *reinterpret_cast<const u32x4*>(arow + k2 * 32);
      aA.q[1] = *reinterpret_cast<const u32x4*>(arow + k2 * 32 + 16);
#pragma unroll
      for (int j = 0; j < 4; ++j)
        bA[j] = *reinterpret_cast<const v16bf*>(wbase +
                                                (size_t)(k2 * 128 + j) * 512);
    }
#pragma unroll
    for (int j = 0; j < 4; ++j) acc[j] = wmma_bf16(aB.v, bB[j], acc[j]);
  }

  const int rb = (lane >> 4) * 8, cl = lane & 15;
#pragma unroll
  for (int j = 0; j < 4; ++j) {
    const int col = (ng * 4 + j) * 16 + cl;
    const float* bp =
        (col < 512) ? bf_ : (col < 1024) ? bi_ : (col < 1536) ? bg_ : bo_;
    const float bias = bp[col & 511];
    float* zp = Zx + (size_t)(mtile * 16 + rb) * NOUT + col;
#pragma unroll
    for (int i = 0; i < 8; ++i) zp[(size_t)i * NOUT] = acc[j][i] + bias;
  }
}

// ---------------------------------------------------------------------------
// Phase 2: persistent recurrent scan. 16 WGs x 4 waves. Each WG owns 32
// h-columns; its Wh slice (128KB bf16) is TDM-preloaded into LDS and stays
// resident. Per step: TDM-broadcast full h (32KB) into LDS, 64 bf16 WMMAs
// per wave (4 gates x 16 K-steps, C init from Zx, double-buffered LDS
// fragment pipeline), gate math with c in registers, h stored bf16 to
// Hseq[t+1], device-scope atomic barrier.
// LDS: 128KB Wh + 32KB h = 160KB of the 320KB WGP pool.
// ---------------------------------------------------------------------------
__global__ void __launch_bounds__(SCAN_THREADS)
lstm_scan_kernel(const float* __restrict__ Zx, const bf16raw* __restrict__ W4P,
                 bf16raw* __restrict__ Hseq, unsigned int* bar) {
  extern __shared__ __attribute__((aligned(128))) bf16raw sh[];
  bf16raw* shW = sh;            // [4 gates][16 kt][2 nt][512]  = 65536 elems
  bf16raw* shH = sh + 65536;    // h row-major [32][512]        = 16384 elems

  const int wg = blockIdx.x;
  const int lane = threadIdx.x & 31;
  const int w4 = threadIdx.x >> 5;
  const int m = w4 >> 1;        // M-tile (batch rows 16m..16m+15)
  const int nt = w4 & 1;        // local N-tile within the 32-col slice

  // Preload this WG's Wh slice: gate g columns g*512 + wg*32 + [0,32)
  // = packed fragments (kt=16..31, ntile = g*32 + wg*2 + {0,1}).
  if (w4 == 0) {
#pragma unroll
    for (int g = 0; g < 4; ++g) {
      tdm_load_2d(shW + g * 16384,
                  W4P + (size_t)(2048 + g * 32 + wg * 2) * 512,
                  /*xdw=*/512, /*rows=*/16, /*stride_dw=*/32768);
    }
    __builtin_amdgcn_s_wait_tensorcnt(0);
  }
  __syncthreads();

  v8f c = (v8f){0,0,0,0,0,0,0,0}; // cell state lives in registers
  const int rb = (lane >> 4) * 8, cl = lane & 15;
  const bf16raw* wb0 = shW + nt * 512 + (lane << 4);

  for (int t = 0; t < SEQ; ++t) {
    // Broadcast h(t) into LDS via TDM (32KB, written by all WGs last step).
    if (w4 == 0) {
      tdm_load_2d(shH, Hseq + (size_t)t * (BATCH * HID),
                  /*xdw=*/8192, /*rows=*/1, /*stride_dw=*/8192);
      __builtin_amdgcn_s_wait_tensorcnt(0);
    }
    __syncthreads();

    // C init = precomputed x-side pre-activations (+bias already folded).
    v8f z[4];
    {
      const float* zp = Zx + ((size_t)t * BATCH + m * 16 + rb) * NOUT +
                        wg * 32 + nt * 16 + cl;
#pragma unroll
      for (int i = 0; i < 8; ++i) {
        z[0][i] = zp[(size_t)i * NOUT];
        z[1][i] = zp[(size_t)i * NOUT + 512];
        z[2][i] = zp[(size_t)i * NOUT + 1024];
        z[3][i] = zp[(size_t)i * NOUT + 1536];
      }
    }

    // z += h @ Wh  (K = 512 -> 16 k-steps, 4 gates share the A fragment).
    // Two-deep double-buffered pipeline over the LDS fragment loads.
    {
      AFrag aA, aB;
      v16bf bA[4], bB[4];
      load_a_frag(aA, shH, m * 16, 0, HID);
#pragma unroll
      for (int g = 0; g < 4; ++g)
        bA[g] = *reinterpret_cast<const v16bf*>(wb0 + g * 16384);
#pragma unroll
      for (int kt = 0; kt < 16; kt += 2) {
        const int k1 = kt + 1;
        load_a_frag(aB, shH, m * 16, k1 * 32, HID);
#pragma unroll
        for (int g = 0; g < 4; ++g)
          bB[g] = *reinterpret_cast<const v16bf*>(wb0 + k1 * 1024 + g * 16384);
#pragma unroll
        for (int g = 0; g < 4; ++g) z[g] = wmma_bf16(aA.v, bA[g], z[g]);

        const int k2 = kt + 2;
        if (k2 < 16) {
          load_a_frag(aA, shH, m * 16, k2 * 32, HID);
#pragma unroll
          for (int g = 0; g < 4; ++g)
            bA[g] = *reinterpret_cast<const v16bf*>(wb0 + k2 * 1024 +
                                                    g * 16384);
        }
#pragma unroll
        for (int g = 0; g < 4; ++g) z[g] = wmma_bf16(aB.v, bB[g], z[g]);
      }
    }

    // Elementwise LSTM cell; write h(t+1) slice bf16 row-major.
    const size_t hbase = (size_t)(t + 1) * (BATCH * HID) +
                         (size_t)(m * 16 + rb) * HID + wg * 32 + nt * 16 + cl;
#pragma unroll
    for (int i = 0; i < 8; ++i) {
      const float f = sigm(z[0][i]);
      const float in = sigm(z[1][i]);
      const float g = tanhf(z[2][i]);
      const float o = sigm(z[3][i]);
      const float cc = f * c[i] + in * g;
      c[i] = cc;
      Hseq[hbase + (size_t)i * HID] = f2bf(o * tanhf(cc));
    }

    // Device-wide release barrier (monotone counter, no reset races).
    __threadfence();
    __syncthreads();
    if (threadIdx.x == 0) {
      __hip_atomic_fetch_add(bar, 1u, __ATOMIC_RELEASE,
                             __HIP_MEMORY_SCOPE_AGENT);
      while (__hip_atomic_load(bar, __ATOMIC_ACQUIRE,
                               __HIP_MEMORY_SCOPE_AGENT) <
             (unsigned)((t + 1) * NWG)) {
        __builtin_amdgcn_s_sleep(1);
      }
    }
    __syncthreads();
  }
}

// ---------------------------------------------------------------------------
// Phase 3: tag logits = Hseq @ W2 + b2, then row-wise log_softmax over 64.
// One wave per M-tile, double-buffered fragment pipeline; D-fragment rows
// live in one 16-lane half, so softmax reductions are shfl_xor(1,2,4,8).
// ---------------------------------------------------------------------------
__global__ void __launch_bounds__(256)
tag_kernel(const bf16raw* __restrict__ Hseq, const bf16raw* __restrict__ W2P,
           const float* b2, float* out) {
  const int wid = (blockIdx.x * 256 + threadIdx.x) >> 5; // 1024 waves
  const int lane = threadIdx.x & 31;
  const int mtile = wid;
  const bf16raw* H = Hseq + BATCH * HID; // skip h(0), rows are t*32+b
  const bf16raw* wbase = W2P + (lane << 4);
  v8f acc[4];
#pragma unroll
  for (int j = 0; j < 4; ++j) acc[j] = (v8f){0,0,0,0,0,0,0,0};

  AFrag aA, aB;
  v16bf bA[4], bB[4];
  load_a_frag(aA, H, mtile * 16, 0, HID);
#pragma unroll
  for (int j = 0; j < 4; ++j)
    bA[j] = *reinterpret_cast<const v16bf*>(wbase + (size_t)j * 512);
#pragma unroll
  for (int kt = 0; kt < 16; kt += 2) {
    const int k1 = kt + 1;
    load_a_frag(aB, H, mtile * 16, k1 * 32, HID);
#pragma unroll
    for (int j = 0; j < 4; ++j)
      bB[j] = *reinterpret_cast<const v16bf*>(wbase +
                                              (size_t)(k1 * 4 + j) * 512);
#pragma unroll
    for (int j = 0; j < 4; ++j) acc[j] = wmma_bf16(aA.v, bA[j], acc[j]);

    const int k2 = kt + 2;
    if (k2 < 16) {
      load_a_frag(aA, H, mtile * 16, k2 * 32, HID);
#pragma unroll
      for (int j = 0; j < 4; ++j)
        bA[j] = *reinterpret_cast<const v16bf*>(wbase +
                                                (size_t)(k2 * 4 + j) * 512);
    }
#pragma unroll
    for (int j = 0; j < 4; ++j) acc[j] = wmma_bf16(aB.v, bB[j], acc[j]);
  }

  const int rb = (lane >> 4) * 8, cl = lane & 15;
#pragma unroll
  for (int i = 0; i < 8; ++i) {
    float v0 = acc[0][i] + b2[cl];
    float v1 = acc[1][i] + b2[16 + cl];
    float v2 = acc[2][i] + b2[32 + cl];
    float v3 = acc[3][i] + b2[48 + cl];
    float mx = fmaxf(fmaxf(v0, v1), fmaxf(v2, v3));
#pragma unroll
    for (int d = 1; d < 16; d <<= 1) mx = fmaxf(mx, __shfl_xor(mx, d, 32));
    float s = __expf(v0 - mx) + __expf(v1 - mx) + __expf(v2 - mx) +
              __expf(v3 - mx);
#pragma unroll
    for (int d = 1; d < 16; d <<= 1) s += __shfl_xor(s, d, 32);
    const float lse = mx + __logf(s);
    const int gr = mtile * 16 + rb + i; // gr = t*32 + b
    const int t = gr >> 5, b = gr & 31;
    float* op = out + ((size_t)b * SEQ + t) * TAGS + cl;
    op[0] = v0 - lse;
    op[16] = v1 - lse;
    op[32] = v2 - lse;
    op[48] = v3 - lse;
  }
}

// ---------------------------------------------------------------------------
// Launch
// ---------------------------------------------------------------------------
extern "C" void kernel_launch(void* const* d_in, const int* in_sizes, int n_in,
                              void* d_out, int out_size, void* d_ws,
                              size_t ws_size, hipStream_t stream) {
  (void)in_sizes; (void)n_in; (void)out_size; (void)ws_size;
  const int* sentence = (const int*)d_in[0];
  const float* E = (const float*)d_in[1];
  const float* Wf = (const float*)d_in[2];
  const float* bf_ = (const float*)d_in[3];
  const float* Wi = (const float*)d_in[4];
  const float* bi_ = (const float*)d_in[5];
  const float* Wg = (const float*)d_in[6];
  const float* bg_ = (const float*)d_in[7];
  const float* Wo = (const float*)d_in[8];
  const float* bo_ = (const float*)d_in[9];
  const float* W2 = (const float*)d_in[10];
  const float* b2 = (const float*)d_in[11];

  char* ws = (char*)d_ws;
  // Workspace layout (all 256B aligned):
  //   [0,256)            barrier counter
  //   Hseq  (S+1)*32*512 bf16 =  16,809,984 B
  //   W4P   32*128*512   bf16 =   4,194,304 B
  //   W2P   16*4*512     bf16 =      65,536 B
  //   Ebf   32000*512    bf16 =  32,768,000 B
  //   Zx    16384*2048   f32  = 134,217,728 B   (~L2 resident)
  unsigned int* bar = (unsigned int*)ws;
  bf16raw* Hseq = (bf16raw*)(ws + 256);
  bf16raw* W4P = (bf16raw*)(ws + 256 + 16809984);
  bf16raw* W2P = (bf16raw*)(ws + 256 + 16809984 + 4194304);
  bf16raw* Ebf = (bf16raw*)(ws + 256 + 16809984 + 4194304 + 65536);
  float* Zx =
      (float*)(ws + 256 + 16809984 + 4194304 + 65536 + 32768000);

  // Allow 160KB dynamic LDS (320KB/WGP on CDNA5). Harmless if already set.
  (void)hipFuncSetAttribute((const void*)lstm_scan_kernel,
                            hipFuncAttributeMaxDynamicSharedMemorySize,
                            160 * 1024);

  init_kernel<<<64, 256, 0, stream>>>(Hseq, bar);
  pack_e_kernel<<<64000, 256, 0, stream>>>(E, Ebf);
  pack_w4_kernel<<<8192, 256, 0, stream>>>(Wf, Wi, Wg, Wo, W4P);
  pack_w2_kernel<<<128, 256, 0, stream>>>(W2, W2P);
  zx_gemm_kernel<<<4096, 256, 0, stream>>>(sentence, Ebf, W4P, bf_, bi_, bg_,
                                           bo_, Zx);
  lstm_scan_kernel<<<NWG, SCAN_THREADS, 160 * 1024, stream>>>(Zx, W4P, Hseq,
                                                              bar);
  tag_kernel<<<128, 256, 0, stream>>>(Hseq, W2P, b2, (float*)d_out);
}
